// GNN_71459665871212
// MI455X (gfx1250) — matrix operand order repaired and checked by reading, not compiled, not run on hardware
//
#include <hip/hip_runtime.h>

#define EMB 300
#define HE  600
#define SLOPE 0.2f

typedef __attribute__((ext_vector_type(16))) _Float16 v16h;
typedef __attribute__((ext_vector_type(8)))  float    v8f;

__device__ inline void atomicMaxF(float* addr, float val) {
  if (val >= 0.0f) atomicMax((int*)addr, __float_as_int(val));
  else             atomicMin((unsigned int*)addr, __float_as_uint(val));
}

__device__ inline float waveRed(float v) {
  #pragma unroll
  for (int o = 16; o > 0; o >>= 1) v += __shfl_down(v, o, 32);
  return v;
}

// ---------------- init: out = bias, amax = -inf, asum = 0 ----------------
__global__ void k_init(float* out, const float* bias, float* amax, float* asum, int N) {
  int idx = blockIdx.x * blockDim.x + threadIdx.x;
  if (idx < N * EMB) out[idx] = bias[idx % EMB];
  if (idx < 2 * N) { amax[idx] = __int_as_float(0xFF800000); asum[idx] = 0.0f; }
}

// ---- pack x (f32) -> f16 A-fragments in contiguous-k (k = 16h + s) A 16x32 order ----
// lane<16: elems 0..15 = K 0..15 ; lane>=16: K 16..31
__global__ void k_packA(const float* x, _Float16* xsw, int N, int KT) {
  int mt = blockIdx.x, kt = blockIdx.y;
  int lane = threadIdx.x;
  int hf = lane >> 4, r = lane & 15;
  int m = mt * 16 + r;
  v16h v;
  #pragma unroll
  for (int i = 0; i < 16; ++i) {
    int k = kt * 32 + 16 * hf + i;
    float f = (m < N && k < EMB) ? x[(size_t)m * EMB + k] : 0.0f;
    v[i] = (_Float16)f;
  }
  *(v16h*)(xsw + (((size_t)mt * KT + kt) * 32 + lane) * 16) = v;
}

// ---- pack W[600,300] as B = W^T (K x N) fragments, contiguous-k (k = 16h + s) B 32x16 order ----
// lanes 0-15: N=lane, K = 0..15 ; lanes 16-31: N=lane-16, K = 16..31
__global__ void k_packB(const float* W, _Float16* wtb, int KT) {
  int nt = blockIdx.x, kt = blockIdx.y;
  int lane = threadIdx.x;
  int hf = lane >> 4, r = lane & 15;
  int n = nt * 16 + r;
  v16h v;
  #pragma unroll
  for (int i = 0; i < 16; ++i) {
    int k = kt * 32 + 16 * hf + i;
    float f = (n < HE && k < EMB) ? W[(size_t)n * EMB + k] : 0.0f;
    v[i] = (_Float16)f;
  }
  *(v16h*)(wtb + (((size_t)nt * KT + kt) * 32 + lane) * 16) = v;
}

// ---------------- WMMA GEMM: h[N][ldh] = x @ W^T + b_lin ----------------
__global__ void k_gemm(const _Float16* xsw, const _Float16* wtb, const float* blin,
                       float* h, int N, int Ntiles, int KT, int ldh) {
  int mt = blockIdx.x;
  int wv = threadIdx.x >> 5;
  int nt = blockIdx.y * (blockDim.x >> 5) + wv;
  if (nt >= Ntiles) return;                 // wave-uniform: EXEC stays all-ones
  int lane = threadIdx.x & 31;
  const v16h* Ap = (const v16h*)xsw + ((size_t)mt * KT) * 32 + lane;
  const v16h* Bp = (const v16h*)wtb + ((size_t)nt * KT) * 32 + lane;
  v8f acc = {};
  for (int kt = 0; kt < KT; ++kt) {
    v16h a = Ap[(size_t)kt * 32];
    v16h b = Bp[(size_t)kt * 32];
    acc = __builtin_amdgcn_wmma_f32_16x16x32_f16(false, a, false, b,
                                                 (short)0, acc, false, false);
  }
  int hf = lane >> 4, r = lane & 15;
  int n = nt * 16 + r;
  float bl = (n < HE) ? blin[n] : 0.0f;
  #pragma unroll
  for (int v = 0; v < 8; ++v) {
    int m = mt * 16 + 8 * hf + v;           // C/D layout: VGPR v -> M = v + 8*hf
    if (m < N) h[(size_t)m * ldh + n] = acc[v] + bl;
  }
}

// ---------------- per-node attention scores s_i, s_j ----------------
__global__ void k_nodeScores(const float* h, const float* att, float* si, float* sj,
                             int N, int ldh) {
  int wv = threadIdx.x >> 5, lane = threadIdx.x & 31;
  int n = blockIdx.x * (blockDim.x >> 5) + wv;
  if (n >= N) return;
  const float* row = h + (size_t)n * ldh;
  float a0 = 0, a1 = 0, b0 = 0, b1 = 0;
  for (int d = lane; d < EMB; d += 32) {
    float h0 = row[d], h1 = row[EMB + d];
    a0 += h0 * att[d];            b0 += h0 * att[EMB + d];
    a1 += h1 * att[2 * EMB + d];  b1 += h1 * att[3 * EMB + d];
  }
  a0 = waveRed(a0); a1 = waveRed(a1); b0 = waveRed(b0); b1 = waveRed(b1);
  if (lane == 0) { si[2*n] = a0; si[2*n+1] = a1; sj[2*n] = b0; sj[2*n+1] = b1; }
}

// ------ 18-entry tables: ctab[t*2+h] = emb1[t].att_j_h ; [12+d*2+h] = emb2[d].att_j_h
__global__ void k_tables(const float* emb1, const float* emb2, const float* att, float* ctab) {
  int idx = blockIdx.x;                     // 0..17
  int lane = threadIdx.x;
  int row = idx >> 1, hd = idx & 1;
  const float* src = (row < 6) ? (emb1 + (size_t)row * HE) : (emb2 + (size_t)(row - 6) * HE);
  const float* aj = att + hd * 2 * EMB + EMB;  // att_j for head hd
  float s = 0.f;
  for (int d = lane; d < EMB; d += 32) s += src[hd * EMB + d] * aj[d];
  s = waveRed(s);
  if (lane == 0) ctab[idx] = s;
}

// ---------------- edge logits + leakyReLU + segment max ----------------
__global__ void k_edgeAlpha(const int* ei, const int* ea, const float* si, const float* sj,
                            const float* ctab, float* raw, float* amax, int E, int N) {
  int e = blockIdx.x * blockDim.x + threadIdx.x;
  if (e >= E + N) return;
  int dst, src, t, dr;
  if (e < E) { dst = ei[e]; src = ei[E + e]; t = ea[2 * e]; dr = ea[2 * e + 1]; }
  else       { dst = src = e - E; t = 4; dr = 0; }
  #pragma unroll
  for (int hd = 0; hd < 2; ++hd) {
    float rA = si[2*dst+hd] + sj[2*src+hd] + ctab[2*t+hd] + ctab[12 + 2*dr + hd];
    rA = rA > 0.f ? rA : SLOPE * rA;
    raw[(size_t)e * 2 + hd] = rA;
    atomicMaxF(&amax[2 * dst + hd], rA);
  }
}

// ---------------- exp(raw - max) + segment sum (in place) ----------------
__global__ void k_edgeExp(const int* ei, float* alp, const float* amax, float* asum,
                          int E, int N) {
  int e = blockIdx.x * blockDim.x + threadIdx.x;
  if (e >= E + N) return;
  int dst = (e < E) ? ei[e] : (e - E);
  #pragma unroll
  for (int hd = 0; hd < 2; ++hd) {
    float a = __expf(alp[(size_t)e * 2 + hd] - amax[2 * dst + hd]);
    alp[(size_t)e * 2 + hd] = a;
    atomicAdd(&asum[2 * dst + hd], a);
  }
}

// -------- weighted scatter with fused head-mean: one block per edge --------
__global__ void k_scatter(const int* ei, const int* ea, const float* h,
                          const float* emb1, const float* emb2,
                          const float* alp, const float* asum, float* out,
                          int E, int N, int ldh) {
  int e = blockIdx.x;
  int dst, src, t, dr;
  if (e < E) { dst = ei[e]; src = ei[E + e]; t = ea[2 * e]; dr = ea[2 * e + 1]; }
  else       { dst = src = e - E; t = 4; dr = 0; }
  float w0 = alp[(size_t)e * 2]     / (asum[2 * dst]     + 1e-16f);
  float w1 = alp[(size_t)e * 2 + 1] / (asum[2 * dst + 1] + 1e-16f);
  int d = threadIdx.x;
  if (d < EMB) {
    float ee0 = emb1[(size_t)t * HE + d]       + emb2[(size_t)dr * HE + d];
    float ee1 = emb1[(size_t)t * HE + EMB + d] + emb2[(size_t)dr * HE + EMB + d];
    const float* row = h + (size_t)src * ldh;
    float v = 0.5f * (w0 * (row[d] + ee0) + w1 * (row[EMB + d] + ee1));
    atomicAdd(&out[(size_t)dst * EMB + d], v);
  }
}

extern "C" void kernel_launch(void* const* d_in, const int* in_sizes, int n_in,
                              void* d_out, int out_size, void* d_ws, size_t ws_size,
                              hipStream_t stream) {
  const float* x    = (const float*)d_in[0];
  const int*   ei   = (const int*)d_in[1];
  const int*   ea   = (const int*)d_in[2];
  const float* W    = (const float*)d_in[3];
  const float* blin = (const float*)d_in[4];
  const float* att  = (const float*)d_in[5];
  const float* bias = (const float*)d_in[6];
  const float* emb1 = (const float*)d_in[7];
  const float* emb2 = (const float*)d_in[8];
  float* out = (float*)d_out;

  int N = in_sizes[0] / EMB;       // 10000 nodes
  int E = in_sizes[1] / 2;         // 160000 edges
  int Etot = E + N;                // + self loops
  int Mtiles = (N + 15) / 16;      // 625
  int Ntiles = (HE + 15) / 16;     // 38
  int ldh = Ntiles * 16;           // 608 (padded h row)
  int KT = (EMB + 31) / 32;        // 10 k-tiles (K padded to 320 with zeros)

  char* ws = (char*)d_ws;
  size_t off = 0;
  auto take = [&](size_t bytes) -> char* {
    off = (off + 255) & ~(size_t)255;
    char* p = ws + off;
    off += bytes;
    return p;
  };
  float*    h    = (float*)   take((size_t)N * ldh * 4);
  _Float16* xsw  = (_Float16*)take((size_t)Mtiles * KT * 32 * 16 * 2);
  _Float16* wtb  = (_Float16*)take((size_t)Ntiles * KT * 32 * 16 * 2);
  float*    si   = (float*)   take((size_t)N * 2 * 4);
  float*    sj   = (float*)   take((size_t)N * 2 * 4);
  float*    ctab = (float*)   take(18 * 4);
  float*    amax = (float*)   take((size_t)N * 2 * 4);
  float*    asum = (float*)   take((size_t)N * 2 * 4);
  float*    alp  = (float*)   take((size_t)Etot * 2 * 4);
  (void)ws_size; (void)n_in; (void)out_size;

  // 1) init output/softmax state
  k_init<<<((size_t)N * EMB + 255) / 256, 256, 0, stream>>>(out, bias, amax, asum, N);
  // 2) pack f16 operands into WMMA fragment layouts
  k_packA<<<dim3(Mtiles, KT), 32, 0, stream>>>(x, xsw, N, KT);
  k_packB<<<dim3(Ntiles, KT), 32, 0, stream>>>(W, wtb, KT);
  // 3) h = x @ W^T + b_lin  (v_wmma_f32_16x16x32_f16)
  k_gemm<<<dim3(Mtiles, (Ntiles + 7) / 8), 256, 0, stream>>>(xsw, wtb, blin, h, N, Ntiles, KT, ldh);
  // 4) per-node score factors + 18-entry edge-attr tables
  k_nodeScores<<<(N + 7) / 8, 256, 0, stream>>>(h, att, si, sj, N, ldh);
  k_tables<<<18, 32, 0, stream>>>(emb1, emb2, att, ctab);
  // 5) segment softmax + scatter
  k_edgeAlpha<<<(Etot + 255) / 256, 256, 0, stream>>>(ei, ea, si, sj, ctab, alp, amax, E, N);
  k_edgeExp<<<(Etot + 255) / 256, 256, 0, stream>>>(ei, alp, amax, asum, E, N);
  k_scatter<<<Etot, 320, 0, stream>>>(ei, ea, h, emb1, emb2, alp, asum, out, E, N, ldh);
}
